// SimpleEdgeModel_82300163326670
// MI455X (gfx1250) — matrix operand order repaired and checked by reading, not compiled, not run on hardware
//
#include <hip/hip_runtime.h>
#include <hip/hip_bf16.h>

typedef _Float16 v4h  __attribute__((ext_vector_type(4)));
typedef _Float16 v8h  __attribute__((ext_vector_type(8)));
typedef _Float16 v16h __attribute__((ext_vector_type(16)));
typedef float    v8f  __attribute__((ext_vector_type(8)));

#define C_IN 64
#define H    128
#define NN   512
#define RS   136   // padded LDS row stride in halves (128 + 8) -> conflict-free b128
#define NW   8     // waves per workgroup in edge kernel
#define MT   2     // M-tiles (16 rows each) per wave: B-fragments feed MT WMMAs

__device__ __forceinline__ float relu_med3(float x) {
  // single v_med3_f32 instead of fmaxf's canonicalize+max pair
  return __builtin_amdgcn_fmed3f(x, 0.0f, __builtin_inff());
}

// ---------------------------------------------------------------------------
// Stage 1: per-node MLP + fold of Wc1:  u = relu(relu(x@Wa+ba)@Wb+bb) @ Wc1
// (the first edge layer is linear in d = h[j]-h[i], so e1 = relu(u[j]-u[i]+bc1))
// ---------------------------------------------------------------------------
__global__ __launch_bounds__(H) void node_mlp_kernel(
    const float* __restrict__ x,  const float* __restrict__ Wa,
    const float* __restrict__ ba, const float* __restrict__ Wb,
    const float* __restrict__ bb, const float* __restrict__ Wc1,
    float* __restrict__ u) {
  const int row = blockIdx.x;           // b*512 + n, 0..1023
  const int t   = threadIdx.x;          // 0..127 (output feature)
  __shared__ float xs[C_IN];
  __shared__ float h1[H];
  __shared__ float h2[H];

  if (t < C_IN) xs[t] = x[row * C_IN + t];
  __syncthreads();

  float a = ba[t];
#pragma unroll 8
  for (int c = 0; c < C_IN; ++c) a = fmaf(xs[c], Wa[c * H + t], a);
  h1[t] = relu_med3(a);
  __syncthreads();

  float s = bb[t];
#pragma unroll 8
  for (int c = 0; c < H; ++c) s = fmaf(h1[c], Wb[c * H + t], s);
  h2[t] = relu_med3(s);
  __syncthreads();

  float uv = 0.f;
#pragma unroll 8
  for (int c = 0; c < H; ++c) uv = fmaf(h2[c], Wc1[c * H + t], uv);
  u[row * H + t] = uv;
}

// ---------------------------------------------------------------------------
// Stage 2: edge MLP via WMMA.
// WG = 256 threads (8 wave32). blockIdx -> (b, i, j-group of 256).
// Wave w handles 32 edges (MT=2 tiles of 16; i fixed, j0 = jg*256 + w*32):
//   act1 = relu(u[j] - u[i] + bc1)          (f16 tile in LDS, direct compute)
//   act2 = relu(act1 @ Wc2 + bc2)           (WMMA; each B-frag feeds 2 tiles)
//   e3   = relu(act2 @ Wc3 + bc3)           (WMMA, kept in regs)
//   out  = e3 @ Wo + bo                     (per-lane dot + shfl_xor reduce)
// ---------------------------------------------------------------------------
__global__ __launch_bounds__(256) void edge_mlp_kernel(
    const float* __restrict__ u,   const float* __restrict__ bc1,
    const float* __restrict__ Wc2, const float* __restrict__ bc2,
    const float* __restrict__ Wc3, const float* __restrict__ bc3,
    const float* __restrict__ Wo,  const float* __restrict__ bo,
    float* __restrict__ out) {
  __shared__ _Float16 sW2[H * RS];                 // Wc2 transposed: [n][k]
  __shared__ _Float16 sW3[H * RS];                 // Wc3 transposed: [n][k]
  __shared__ _Float16 sAct[NW * 2 * (MT * 16) * RS]; // per-wave act1/act2, 32x128

  const int tid = threadIdx.x;

  // Stage transposed f16 weights into LDS (coalesced global reads).
  for (int idx = tid; idx < H * H; idx += 256) {
    const int k = idx >> 7, n = idx & (H - 1);
    sW2[n * RS + k] = (_Float16)Wc2[idx];
    sW3[n * RS + k] = (_Float16)Wc3[idx];
  }

  const int jg = blockIdx.x & 1;                 // j-group of 256
  const int i  = (blockIdx.x >> 1) & (NN - 1);
  const int b  = blockIdx.x >> 10;

  const int wave = tid >> 5;
  const int lane = tid & 31;
  const int nl   = lane & 15;   // N within 16-tile / M for A rows
  const int g    = lane >> 4;   // half-wave group
  const int j0   = jg * 256 + wave * 32;

  const float* ub = u + (size_t)b * NN * H;

  // Per-lane constants.
  const float4 ui4 = *(const float4*)(ub + i * H + lane * 4);
  const float4 b14 = *(const float4*)(bc1 + lane * 4);
  float bc2r[8], bc3r[8], wor[8];
#pragma unroll
  for (int nt = 0; nt < 8; ++nt) {
    bc2r[nt] = bc2[nl + 16 * nt];
    bc3r[nt] = bc3[nl + 16 * nt];
    wor[nt]  = Wo[nl + 16 * nt];
  }

  _Float16* act1 = sAct + wave * (2 * MT * 16 * RS);
  _Float16* act2 = act1 + MT * 16 * RS;

  // Layer 1 (free matmul): act1[m][k] = relu(u[j0+m][k] - u[i][k] + bc1[k]).
#pragma unroll 4
  for (int r = 0; r < MT * 16; ++r) {
    const float4 uj = *(const float4*)(ub + (size_t)(j0 + r) * H + lane * 4);
    v4h v;
    v.x = (_Float16)relu_med3(uj.x - ui4.x + b14.x);
    v.y = (_Float16)relu_med3(uj.y - ui4.y + b14.y);
    v.z = (_Float16)relu_med3(uj.z - ui4.z + b14.z);
    v.w = (_Float16)relu_med3(uj.w - ui4.w + b14.w);
    *(v4h*)(act1 + r * RS + lane * 4) = v;
  }
  __syncthreads();   // weights + act tiles visible

  // ---- Layer 2: act2 = relu(act1 @ Wc2 + bc2) ----
  {
    // A fragments (ISA 16-bit A layout): lane holds row M=nl,
    // halves h<8 -> K = 32kc + 8g + h ; h>=8 -> K = 32kc + 8g + 16 + (h-8).
    v16h A[MT][4];
#pragma unroll
    for (int t = 0; t < MT; ++t) {
      const _Float16* aRow = act1 + (t * 16 + nl) * RS + 8 * g;
#pragma unroll
      for (int kc = 0; kc < 4; ++kc) {
        const v8h lo = *(const v8h*)(aRow + 32 * kc);
        const v8h hi = *(const v8h*)(aRow + 32 * kc + 16);
        A[t][kc] = __builtin_shufflevector(lo, hi, 0,1,2,3,4,5,6,7,8,9,10,11,12,13,14,15);
      }
    }
#pragma unroll
    for (int nt = 0; nt < 8; ++nt) {
      const float bv = bc2r[nt];
      v8f acc[MT];
#pragma unroll
      for (int t = 0; t < MT; ++t) acc[t] = (v8f){bv, bv, bv, bv, bv, bv, bv, bv};
      const _Float16* wrow = sW2 + (nl + 16 * nt) * RS + 8 * g;
#pragma unroll
      for (int kc = 0; kc < 4; ++kc) {
        const v8h lo = *(const v8h*)(wrow + 32 * kc);
        const v8h hi = *(const v8h*)(wrow + 32 * kc + 16);
        const v16h B = __builtin_shufflevector(lo, hi, 0,1,2,3,4,5,6,7,8,9,10,11,12,13,14,15);
#pragma unroll
        for (int t = 0; t < MT; ++t)
          acc[t] = __builtin_amdgcn_wmma_f32_16x16x32_f16(false, A[t][kc], false, B,
                                                          (short)0, acc[t], false, false);
      }
      // D layout: VGPR r -> (M = r + 8g, N = nl + 16nt). relu -> f16 -> LDS.
#pragma unroll
      for (int t = 0; t < MT; ++t) {
        _Float16* dst = act2 + (t * 16 + 8 * g) * RS + (nl + 16 * nt);
#pragma unroll
        for (int r = 0; r < 8; ++r)
          dst[r * RS] = (_Float16)relu_med3(acc[t][r]);
      }
    }
  }
  __syncthreads();

  // ---- Layer 3 + output projection fused ----
  {
    v16h A[MT][4];
#pragma unroll
    for (int t = 0; t < MT; ++t) {
      const _Float16* aRow = act2 + (t * 16 + nl) * RS + 8 * g;
#pragma unroll
      for (int kc = 0; kc < 4; ++kc) {
        const v8h lo = *(const v8h*)(aRow + 32 * kc);
        const v8h hi = *(const v8h*)(aRow + 32 * kc + 16);
        A[t][kc] = __builtin_shufflevector(lo, hi, 0,1,2,3,4,5,6,7,8,9,10,11,12,13,14,15);
      }
    }
    float part[MT][8];
#pragma unroll
    for (int t = 0; t < MT; ++t)
#pragma unroll
      for (int r = 0; r < 8; ++r) part[t][r] = 0.f;

#pragma unroll
    for (int nt = 0; nt < 8; ++nt) {
      const float bv = bc3r[nt];
      v8f acc[MT];
#pragma unroll
      for (int t = 0; t < MT; ++t) acc[t] = (v8f){bv, bv, bv, bv, bv, bv, bv, bv};
      const _Float16* wrow = sW3 + (nl + 16 * nt) * RS + 8 * g;
#pragma unroll
      for (int kc = 0; kc < 4; ++kc) {
        const v8h lo = *(const v8h*)(wrow + 32 * kc);
        const v8h hi = *(const v8h*)(wrow + 32 * kc + 16);
        const v16h B = __builtin_shufflevector(lo, hi, 0,1,2,3,4,5,6,7,8,9,10,11,12,13,14,15);
#pragma unroll
        for (int t = 0; t < MT; ++t)
          acc[t] = __builtin_amdgcn_wmma_f32_16x16x32_f16(false, A[t][kc], false, B,
                                                          (short)0, acc[t], false, false);
      }
      const float w = wor[nt];
#pragma unroll
      for (int t = 0; t < MT; ++t)
#pragma unroll
        for (int r = 0; r < 8; ++r)
          part[t][r] = fmaf(relu_med3(acc[t][r]), w, part[t][r]);
    }
    // Reduce over the 16 lanes of each half-wave (features nl=0..15).
#pragma unroll
    for (int m = 1; m <= 8; m <<= 1)
#pragma unroll
      for (int t = 0; t < MT; ++t)
#pragma unroll
        for (int r = 0; r < 8; ++r)
          part[t][r] += __shfl_xor(part[t][r], m, 32);

    if (nl == 0) {
      const float bov = bo[0];
      const size_t base = ((size_t)b * NN + i) * NN + j0;
#pragma unroll
      for (int t = 0; t < MT; ++t)
#pragma unroll
        for (int r = 0; r < 8; ++r)
          out[base + t * 16 + r + 8 * g] = part[t][r] + bov;
    }
  }
}

// ---------------------------------------------------------------------------
extern "C" void kernel_launch(void* const* d_in, const int* in_sizes, int n_in,
                              void* d_out, int out_size, void* d_ws, size_t ws_size,
                              hipStream_t stream) {
  (void)in_sizes; (void)n_in; (void)out_size; (void)ws_size;
  const float* x   = (const float*)d_in[0];
  const float* Wa  = (const float*)d_in[1];
  const float* ba  = (const float*)d_in[2];
  const float* Wb  = (const float*)d_in[3];
  const float* bb  = (const float*)d_in[4];
  const float* Wc1 = (const float*)d_in[5];
  const float* bc1 = (const float*)d_in[6];
  const float* Wc2 = (const float*)d_in[7];
  const float* bc2 = (const float*)d_in[8];
  const float* Wc3 = (const float*)d_in[9];
  const float* bc3 = (const float*)d_in[10];
  const float* Wo  = (const float*)d_in[11];
  const float* bo  = (const float*)d_in[12];
  float* out = (float*)d_out;
  float* u   = (float*)d_ws;             // [2][512][128] fp32 = 512 KB

  node_mlp_kernel<<<2 * NN, H, 0, stream>>>(x, Wa, ba, Wb, bb, Wc1, u);
  edge_mlp_kernel<<<2 * NN * 2, 256, 0, stream>>>(u, bc1, Wc2, bc2, Wc3, bc3,
                                                  Wo, bo, out);
}
